// AttentionBlock_55130200211784
// MI455X (gfx1250) — matrix operand order repaired and checked by reading, not compiled, not run on hardware
//
#include <hip/hip_runtime.h>

// ---------------- constants ----------------
constexpr int Bn = 16, Cc = 512, Ll = 2048, Hh = 8, Dh = 64;
constexpr int C2 = 1024;                 // 2*C
constexpr int BL = Bn * Ll;              // 32768 rows
constexpr float EPS = 1e-5f;

// ---------------- types ----------------
typedef __bf16 v16bf __attribute__((ext_vector_type(16)));
typedef float  v8f   __attribute__((ext_vector_type(8)));
typedef int    v4i   __attribute__((ext_vector_type(4)));

struct U4 { unsigned int x, y, z, w; };           // POD 16-byte chunk
union Frag { v16bf v; U4 u[2]; };                 // 32 bytes: one WMMA A/B fragment

// CDNA5 async memory->LDS path (gated: falls back to register staging if absent)
#if defined(__has_builtin)
#if __has_builtin(__builtin_amdgcn_global_load_async_to_lds_b128) && \
    __has_builtin(__builtin_amdgcn_s_wait_asynccnt)
#define USE_ASYNC_LDS 1
#endif
#endif
#ifndef USE_ASYNC_LDS
#define USE_ASYNC_LDS 0
#endif

#if USE_ASYNC_LDS
typedef __attribute__((address_space(1))) v4i* gptr4;   // global v4i*
typedef __attribute__((address_space(3))) v4i* lptr4;   // LDS v4i*
#endif

__device__ __forceinline__ unsigned short f2bf(float f) {
  unsigned int u = __float_as_uint(f);
  u += 0x7FFFu + ((u >> 16) & 1u);                // round-to-nearest-even
  return (unsigned short)(u >> 16);
}
__device__ __forceinline__ float bf2f(unsigned short h) {
  return __uint_as_float(((unsigned int)h) << 16);
}

// ---------------- weight convert: W[K][N] f32 -> Wt[N][K] bf16 ----------------
__global__ void k_transpose_w_bf16(const float* __restrict__ W,
                                   unsigned short* __restrict__ Wt, int Kd, int Nd) {
  long long i = (long long)blockIdx.x * blockDim.x + threadIdx.x;
  long long total = (long long)Kd * Nd;
  if (i >= total) return;
  int k = (int)(i / Nd), n = (int)(i % Nd);
  Wt[(long long)n * Kd + k] = f2bf(W[i]);
}

// ---------------- instance-norm stats over L per (b,c) ----------------
template<bool CH_MAJOR>
__global__ void k_stats(const float* __restrict__ x, float* __restrict__ mean,
                        float* __restrict__ rstd) {
  int bc = blockIdx.x;
  const float* p; int stride;
  if (CH_MAJOR) { p = x + (long long)bc * Ll; stride = 1; }
  else { int b = bc / Cc, c = bc % Cc; p = x + (long long)b * Ll * Cc + c; stride = Cc; }
  float s = 0.f, q = 0.f;
  for (int l = threadIdx.x; l < Ll; l += blockDim.x) {
    float v = p[(long long)l * stride]; s += v; q += v * v;
  }
  __shared__ float sh_s[256], sh_q[256];
  int t = threadIdx.x;
  sh_s[t] = s; sh_q[t] = q; __syncthreads();
  for (int o = 128; o > 0; o >>= 1) {
    if (t < o) { sh_s[t] += sh_s[t + o]; sh_q[t] += sh_q[t + o]; }
    __syncthreads();
  }
  if (t == 0) {
    float m = sh_s[0] / (float)Ll;
    float var = sh_q[0] / (float)Ll - m * m;     // biased var (matches jnp.var)
    mean[bc] = m; rstd[bc] = rsqrtf(var + EPS);
  }
}

// ---------------- normalize + transpose (B,C,L) f32 -> (B,L,C) bf16 ----------------
__global__ void k_norm1_transpose(const float* __restrict__ x, const float* __restrict__ mean,
                                  const float* __restrict__ rstd, const float* __restrict__ g,
                                  const float* __restrict__ be, unsigned short* __restrict__ out) {
  long long i = (long long)blockIdx.x * blockDim.x + threadIdx.x;   // over (b,c,l)
  if (i >= (long long)Bn * Cc * Ll) return;
  int l = (int)(i % Ll); long long r = i / Ll;
  int c = (int)(r % Cc); int b = (int)(r / Cc);
  int bc = b * Cc + c;
  float v = (x[i] - mean[bc]) * rstd[bc] * g[c] + be[c];
  out[((long long)b * Ll + l) * Cc + c] = f2bf(v);
}

// ---------------- plain transpose (B,C,L) f32 -> (B,L,C) bf16 ----------------
__global__ void k_transpose_bf16(const float* __restrict__ x, unsigned short* __restrict__ out) {
  long long i = (long long)blockIdx.x * blockDim.x + threadIdx.x;
  if (i >= (long long)Bn * Cc * Ll) return;
  int l = (int)(i % Ll); long long r = i / Ll;
  int c = (int)(r % Cc); int b = (int)(r / Cc);
  out[((long long)b * Ll + l) * Cc + c] = f2bf(x[i]);
}

// ---------------- norm2: (B,L,C) f32 -> (B,L,C) bf16, stats per (b,c) ----------------
__global__ void k_norm2(const float* __restrict__ a, const float* __restrict__ mean,
                        const float* __restrict__ rstd, const float* __restrict__ g,
                        const float* __restrict__ be, unsigned short* __restrict__ out) {
  long long i = (long long)blockIdx.x * blockDim.x + threadIdx.x;   // over (b,l,c)
  if (i >= (long long)BL * Cc) return;
  int c = (int)(i % Cc);
  int b = (int)(i / ((long long)Ll * Cc));
  int bc = b * Cc + c;
  out[i] = f2bf((a[i] - mean[bc]) * rstd[bc] * g[c] + be[c]);
}

// ---------------- windowed attention: one wave per (b,l,h) ----------------
__global__ void k_window_attn(const unsigned short* __restrict__ Q,
                              const unsigned short* __restrict__ Kk,
                              const unsigned short* __restrict__ V,
                              unsigned short* __restrict__ O) {
  int gt = blockIdx.x * blockDim.x + threadIdx.x;
  int wave = gt >> 5;
  int lane = threadIdx.x & 31;
  int h = wave % Hh; int t2 = wave / Hh;
  int l = t2 % Ll;   int b  = t2 / Ll;
  long long row = ((long long)b * Ll + l) * Cc + h * Dh;
  float q0 = bf2f(Q[row + lane]);
  float q1 = bf2f(Q[row + lane + 32]);
  float sc[3];
#pragma unroll
  for (int o = 0; o < 3; ++o) {
    int kl = l + o - 1;
    bool valid = (kl >= 0) && (kl < Ll);
    float d = 0.f;
    if (valid) {
      long long rk = ((long long)b * Ll + kl) * Cc + h * Dh;
      d = q0 * bf2f(Kk[rk + lane]) + q1 * bf2f(Kk[rk + lane + 32]);
    }
#pragma unroll
    for (int m = 16; m >= 1; m >>= 1) d += __shfl_xor(d, m, 32);
    sc[o] = valid ? d * 0.125f : -1e9f;          // 1/sqrt(Dh)=0.125
  }
  float mx = fmaxf(sc[0], fmaxf(sc[1], sc[2]));
  float e0 = __expf(sc[0] - mx), e1 = __expf(sc[1] - mx), e2 = __expf(sc[2] - mx);
  float inv = 1.f / (e0 + e1 + e2);
  float wgt[3] = { e0 * inv, e1 * inv, e2 * inv };
  float o0 = 0.f, o1 = 0.f;
#pragma unroll
  for (int o = 0; o < 3; ++o) {
    int kl = l + o - 1;
    if (kl >= 0 && kl < Ll) {
      long long rv = ((long long)b * Ll + kl) * Cc + h * Dh;
      o0 += wgt[o] * bf2f(V[rv + lane]);
      o1 += wgt[o] * bf2f(V[rv + lane + 32]);
    }
  }
  O[row + lane]      = f2bf(o0);
  O[row + lane + 32] = f2bf(o1);
}

// ---------------- WMMA bf16 GEMM: D[M,N] = A[M,K] * Wt[N,K]^T + bias ----------------
// Block tile 128x128, BK=64 (two WMMA K-steps per stage), double-buffered LDS,
// software pipelined: stage tile t+1 (async DMA if available) while computing tile t.
// 8 waves -> each wave owns 64x32 (4x2 frags of 16x16).
enum { EPI_F32 = 0, EPI_BF16 = 1, EPI_GELU_BF16 = 2, EPI_RESID = 3 };

template<int EPI>
__global__ __launch_bounds__(256)
void k_gemm_bf16_wmma(const unsigned short* __restrict__ A,
                      const unsigned short* __restrict__ Wt,
                      const float* __restrict__ bias, void* __restrict__ outv,
                      const float* __restrict__ resid, int M, int N, int K) {
  // tile: 128 rows x 64 bf16 (128 bytes per row), double buffered
  __shared__ alignas(16) unsigned short As[2][128 * 64];
  __shared__ alignas(16) unsigned short Bs[2][128 * 64];
  const int tid   = threadIdx.x;
  const int lane  = tid & 31;
  const int wave  = tid >> 5;
  const int waveM = wave >> 2;                  // 0..1  -> 64-row slab
  const int waveN = wave & 3;                   // 0..3  -> 32-col slab
  const int mBase = blockIdx.y * 128;
  const int nBase = blockIdx.x * 128;
  const int half  = lane >> 4;
  const int r16   = lane & 15;

  v8f acc[4][2];
#pragma unroll
  for (int i = 0; i < 4; ++i)
#pragma unroll
    for (int j = 0; j < 2; ++j)
#pragma unroll
      for (int e = 0; e < 8; ++e) acc[i][j][e] = 0.f;

  // staging: 16 KiB per matrix per tile; each thread moves 4 chunks of 16 B.
  // chunk c: byte index li = c*4096 + tid*16 ; row = li>>7 ; col-byte = li&127
  const int numT = K >> 6;

#if USE_ASYNC_LDS
  auto stage_async = [&](int t, int buf) {
#pragma unroll
    for (int c = 0; c < 4; ++c) {
      int li   = c * 4096 + tid * 16;
      int row  = li >> 7;
      int cole = (li & 127) >> 1;               // element offset in row
      const unsigned short* ga = A  + (long long)(mBase + row) * K + t * 64 + cole;
      const unsigned short* gb = Wt + (long long)(nBase + row) * K + t * 64 + cole;
      __builtin_amdgcn_global_load_async_to_lds_b128(
          (gptr4)ga, (lptr4)(&As[buf][row * 64 + cole]), 0, 0);
      __builtin_amdgcn_global_load_async_to_lds_b128(
          (gptr4)gb, (lptr4)(&Bs[buf][row * 64 + cole]), 0, 0);
    }
  };
#else
  U4 ra[4], rb[4];
  auto load_regs = [&](int t) {
#pragma unroll
    for (int c = 0; c < 4; ++c) {
      int li   = c * 4096 + tid * 16;
      int row  = li >> 7;
      int cole = (li & 127) >> 1;
      ra[c] = *(const U4*)(A  + (long long)(mBase + row) * K + t * 64 + cole);
      rb[c] = *(const U4*)(Wt + (long long)(nBase + row) * K + t * 64 + cole);
    }
  };
  auto store_regs = [&](int buf) {
#pragma unroll
    for (int c = 0; c < 4; ++c) {
      int li   = c * 4096 + tid * 16;
      int row  = li >> 7;
      int cole = (li & 127) >> 1;
      *(U4*)(&As[buf][row * 64 + cole]) = ra[c];
      *(U4*)(&Bs[buf][row * 64 + cole]) = rb[c];
    }
  };
#endif

  // prologue: stage tile 0 into buffer 0
#if USE_ASYNC_LDS
  stage_async(0, 0);
  __builtin_amdgcn_s_wait_asynccnt(0);
#else
  load_regs(0);
  store_regs(0);
#endif
  __syncthreads();

  for (int t = 0; t < numT; ++t) {
    const int cur = t & 1;
    const int nxt = cur ^ 1;
#if USE_ASYNC_LDS
    if (t + 1 < numT) stage_async(t + 1, nxt);  // DMA next tile while computing
#else
    if (t + 1 < numT) load_regs(t + 1);         // issue global loads early
#endif

#pragma unroll
    for (int s = 0; s < 2; ++s) {               // two 16x16x32 K-steps per tile
      Frag a[4], bfr[2];
#pragma unroll
      for (int i = 0; i < 4; ++i) {             // A frag: K = {half*8..+8} U {16+half*8..+8}
        int base = (waveM * 64 + i * 16 + r16) * 64 + s * 32;
        a[i].u[0] = *(const U4*)(&As[cur][base + half * 8]);
        a[i].u[1] = *(const U4*)(&As[cur][base + 16 + half * 8]);
      }
#pragma unroll
      for (int j = 0; j < 2; ++j) {             // B frag: K = {half*8..+8} U {16+half*8..+8}
        int base = (waveN * 32 + j * 16 + r16) * 64 + s * 32;
        bfr[j].u[0] = *(const U4*)(&Bs[cur][base + half * 8]);
        bfr[j].u[1] = *(const U4*)(&Bs[cur][base + 16 + half * 8]);
      }
#pragma unroll
      for (int i = 0; i < 4; ++i)
#pragma unroll
        for (int j = 0; j < 2; ++j)
          acc[i][j] = __builtin_amdgcn_wmma_f32_16x16x32_bf16(
              false, a[i].v, false, bfr[j].v, (short)0, acc[i][j], false, false);
    }

#if USE_ASYNC_LDS
    if (t + 1 < numT) __builtin_amdgcn_s_wait_asynccnt(0);
#else
    if (t + 1 < numT) store_regs(nxt);
#endif
    __syncthreads();
  }

  // epilogue: D element (v, half): row = v + 8*half, col = lane&15
#pragma unroll
  for (int i = 0; i < 4; ++i) {
#pragma unroll
    for (int j = 0; j < 2; ++j) {
      int col = nBase + waveN * 32 + j * 16 + r16;
      float bvl = bias[col];
#pragma unroll
      for (int v = 0; v < 8; ++v) {
        int row = mBase + waveM * 64 + i * 16 + half * 8 + v;
        float val = acc[i][j][v] + bvl;
        if (EPI == EPI_F32) {
          ((float*)outv)[(long long)row * N + col] = val;
        } else if (EPI == EPI_BF16) {
          ((unsigned short*)outv)[(long long)row * N + col] = f2bf(val);
        } else if (EPI == EPI_GELU_BF16) {
          float gv = val * (1.f / (1.f + __expf(-1.702f * val)));   // QuickGELU
          ((unsigned short*)outv)[(long long)row * N + col] = f2bf(gv);
        } else {                                   // residual + transpose to (B,C,L)
          int b = row >> 11;                       // row / L
          int l = row & (Ll - 1);
          long long oi = ((long long)b * Cc + col) * Ll + l;
          ((float*)outv)[oi] = resid[oi] + val;
        }
      }
    }
  }
}

// ---------------- host ----------------
extern "C" void kernel_launch(void* const* d_in, const int* in_sizes, int n_in,
                              void* d_out, int out_size, void* d_ws, size_t ws_size,
                              hipStream_t stream) {
  const float* q   = (const float*)d_in[0];
  const float* x   = (const float*)d_in[1];
  const float* g1  = (const float*)d_in[2];
  const float* b1  = (const float*)d_in[3];
  const float* Wq  = (const float*)d_in[4];
  const float* bq  = (const float*)d_in[5];
  const float* Wk  = (const float*)d_in[6];
  const float* bk  = (const float*)d_in[7];
  const float* Wv  = (const float*)d_in[8];
  const float* bv  = (const float*)d_in[9];
  const float* Wo  = (const float*)d_in[10];
  const float* bo  = (const float*)d_in[11];
  const float* g2  = (const float*)d_in[12];
  const float* b2  = (const float*)d_in[13];
  const float* Wfc = (const float*)d_in[14];
  const float* bfc = (const float*)d_in[15];
  const float* Wpr = (const float*)d_in[16];
  const float* bpr = (const float*)d_in[17];

  char* wptr = (char*)d_ws;
  auto alloc = [&](size_t bytes) -> void* {
    void* p = (void*)wptr;
    wptr += (bytes + 255) & ~(size_t)255;
    return p;
  };

  unsigned short* WqT  = (unsigned short*)alloc((size_t)Cc * Cc * 2);
  unsigned short* WkT  = (unsigned short*)alloc((size_t)Cc * Cc * 2);
  unsigned short* WvT  = (unsigned short*)alloc((size_t)Cc * Cc * 2);
  unsigned short* WoT  = (unsigned short*)alloc((size_t)Cc * Cc * 2);
  unsigned short* WfcT = (unsigned short*)alloc((size_t)Cc * C2 * 2);
  unsigned short* WprT = (unsigned short*)alloc((size_t)C2 * Cc * 2);
  float* mean1 = (float*)alloc((size_t)Bn * Cc * 4);
  float* rstd1 = (float*)alloc((size_t)Bn * Cc * 4);
  float* mean2 = (float*)alloc((size_t)Bn * Cc * 4);
  float* rstd2 = (float*)alloc((size_t)Bn * Cc * 4);
  unsigned short* qbf  = (unsigned short*)alloc((size_t)BL * Cc * 2);
  unsigned short* htbf = (unsigned short*)alloc((size_t)BL * Cc * 2);
  unsigned short* Qbf  = (unsigned short*)alloc((size_t)BL * Cc * 2);
  unsigned short* Kbf  = (unsigned short*)alloc((size_t)BL * Cc * 2);
  unsigned short* Vbf  = (unsigned short*)alloc((size_t)BL * Cc * 2);
  float* ao            = (float*)alloc((size_t)BL * Cc * 4);
  unsigned short* ubf  = (unsigned short*)alloc((size_t)BL * C2 * 2);
  unsigned short* obf  = qbf;    // alias: q-bf16 dead after Q projection
  unsigned short* h2bf = htbf;   // alias: ht-bf16 dead after K/V projections

  auto cdiv = [](long long a, long long b) { return (int)((a + b - 1) / b); };
  const long long NE = (long long)Bn * Cc * Ll;   // elements of a (B,C,L) tensor

  // 1) weights -> bf16, N-major
  k_transpose_w_bf16<<<cdiv((long long)Cc * Cc, 256), 256, 0, stream>>>(Wq, WqT, Cc, Cc);
  k_transpose_w_bf16<<<cdiv((long long)Cc * Cc, 256), 256, 0, stream>>>(Wk, WkT, Cc, Cc);
  k_transpose_w_bf16<<<cdiv((long long)Cc * Cc, 256), 256, 0, stream>>>(Wv, WvT, Cc, Cc);
  k_transpose_w_bf16<<<cdiv((long long)Cc * Cc, 256), 256, 0, stream>>>(Wo, WoT, Cc, Cc);
  k_transpose_w_bf16<<<cdiv((long long)Cc * C2, 256), 256, 0, stream>>>(Wfc, WfcT, Cc, C2);
  k_transpose_w_bf16<<<cdiv((long long)C2 * Cc, 256), 256, 0, stream>>>(Wpr, WprT, C2, Cc);

  // 2) norm1 stats + normalize/transpose, q transpose
  k_stats<true><<<Bn * Cc, 256, 0, stream>>>(x, mean1, rstd1);
  k_norm1_transpose<<<cdiv(NE, 256), 256, 0, stream>>>(x, mean1, rstd1, g1, b1, htbf);
  k_transpose_bf16<<<cdiv(NE, 256), 256, 0, stream>>>(q, qbf);

  // 3) Q/K/V projections (bf16 out)
  dim3 gC(Cc / 128, BL / 128);   // (4, 256)
  k_gemm_bf16_wmma<EPI_BF16><<<gC, 256, 0, stream>>>(qbf,  WqT, bq, Qbf, nullptr, BL, Cc, Cc);
  k_gemm_bf16_wmma<EPI_BF16><<<gC, 256, 0, stream>>>(htbf, WkT, bk, Kbf, nullptr, BL, Cc, Cc);
  k_gemm_bf16_wmma<EPI_BF16><<<gC, 256, 0, stream>>>(htbf, WvT, bv, Vbf, nullptr, BL, Cc, Cc);

  // 4) windowed attention (one wave per (b,l,h)) -> obf (aliases qbf)
  k_window_attn<<<(Bn * Ll * Hh) / 8, 256, 0, stream>>>(Qbf, Kbf, Vbf, obf);

  // 5) output projection (f32 out), norm2 stats, normalize -> h2bf
  k_gemm_bf16_wmma<EPI_F32><<<gC, 256, 0, stream>>>(obf, WoT, bo, ao, nullptr, BL, Cc, Cc);
  k_stats<false><<<Bn * Cc, 256, 0, stream>>>(ao, mean2, rstd2);
  k_norm2<<<cdiv((long long)BL * Cc, 256), 256, 0, stream>>>(ao, mean2, rstd2, g2, b2, h2bf);

  // 6) MLP: fc + QuickGELU (bf16 out), then proj + residual + transpose to (B,C,L)
  dim3 gF(C2 / 128, BL / 128);   // (8, 256)
  k_gemm_bf16_wmma<EPI_GELU_BF16><<<gF, 256, 0, stream>>>(h2bf, WfcT, bfc, ubf, nullptr, BL, C2, Cc);
  k_gemm_bf16_wmma<EPI_RESID><<<gC, 256, 0, stream>>>(ubf, WprT, bpr, d_out, x, BL, Cc, C2);
}
